// Model_12807592476901
// MI455X (gfx1250) — matrix operand order repaired and checked by reading, compiled, not run on hardware
//
#include <hip/hip_runtime.h>
#include <hip/hip_bf16.h>

// ---------------- model constants ----------------
#define DM    1024          // D_MODEL
#define DI    2048          // D_INNER
#define DS    64            // D_STATE
#define HD    128           // HEADDIM
#define NH    16            // NHEADS
#define NL    4             // N_LAYERS
#define CDIM  2176          // D_INNER + 2*D_STATE
#define DIP   4240          // 2*D_INNER + 2*D_STATE + NHEADS
#define BSZ   2
#define SEQL  4162          // 1 + 64 + 1 + 4096
#define MTOK  (BSZ*SEQL)    // 8324 tokens

typedef __bf16 bf16_t;
typedef __attribute__((ext_vector_type(16))) __bf16 v16bf;
typedef __attribute__((ext_vector_type(8)))  float  v8f;

__device__ __forceinline__ float siluf(float v){ return v / (1.f + __expf(-v)); }

// ---------------- f32 -> bf16 convert ----------------
__global__ void k_f2bf(const float* __restrict__ src, bf16_t* __restrict__ dst, int n){
    int i = blockIdx.x * 256 + threadIdx.x;
    if (i < n) dst[i] = (bf16_t)src[i];
}

// ---------------- embedding assembly ----------------
__global__ void k_embed(const float* __restrict__ im8, const float* __restrict__ frw,
                        const float* __restrict__ frb, const float* __restrict__ s0,
                        const float* __restrict__ s1,  const float* __restrict__ pe,
                        float* __restrict__ x){
    int tok = blockIdx.x;
    int b = tok / SEQL, t = tok % SEQL;
    for (int i = threadIdx.x; i < DM; i += 256) {
        float v;
        if (t == 0)       v = s0[i];
        else if (t == 65) v = s1[i];
        else {
            int p; float add = 0.f;
            if (t <= 64) p = t - 1;
            else { int q = t - 66; int r = q >> 6, c = q & 63;
                   p = (r >> 3) * 8 + (c >> 3); add = pe[(size_t)q * DM + i]; }
            const float* px = im8 + (size_t)(b * 64 + p) * 3;
            v = px[0]*frw[i] + px[1]*frw[DM+i] + px[2]*frw[2*DM+i] + frb[i] + add;
        }
        x[(size_t)tok * DM + i] = v;
    }
}

// ---------------- LayerNorm -> bf16 ----------------
__global__ void k_ln(const float* __restrict__ x, const float* __restrict__ w,
                     const float* __restrict__ bia, bf16_t* __restrict__ u){
    __shared__ float red[256];
    int tok = blockIdx.x;
    const float* xr = x + (size_t)tok * DM;
    float v[4]; float s = 0.f;
    #pragma unroll
    for (int i = 0; i < 4; i++){ v[i] = xr[threadIdx.x + i*256]; s += v[i]; }
    red[threadIdx.x] = s; __syncthreads();
    for (int o = 128; o > 0; o >>= 1){ if (threadIdx.x < o) red[threadIdx.x] += red[threadIdx.x+o]; __syncthreads(); }
    float mean = red[0] / DM; __syncthreads();
    s = 0.f;
    #pragma unroll
    for (int i = 0; i < 4; i++){ float d = v[i] - mean; s += d*d; }
    red[threadIdx.x] = s; __syncthreads();
    for (int o = 128; o > 0; o >>= 1){ if (threadIdx.x < o) red[threadIdx.x] += red[threadIdx.x+o]; __syncthreads(); }
    float rstd = rsqrtf(red[0] / DM + 1e-5f);
    #pragma unroll
    for (int i = 0; i < 4; i++){
        int d = threadIdx.x + i*256;
        u[(size_t)tok * DM + d] = (bf16_t)((v[i] - mean) * rstd * w[d] + bia[d]);
    }
}

// ---------------- WMMA bf16 GEMM: C[M,N] (+)= A[M,K] * W[N,K]^T ----------------
// 256 threads = 8 waves (4x2); block tile 128(M) x 64(N); wave tile 32x32
// (2x2 of 16x16) -> 8 v_wmma per 64-deep K stage. Register double buffering.
__global__ __launch_bounds__(256)
void k_gemm_bf16(const bf16_t* __restrict__ A, int lda,
                 const bf16_t* __restrict__ W, int ldw,
                 float* __restrict__ C, int ldc,
                 int M, int N, int K, int residual){
    __shared__ __align__(16) bf16_t sA[128 * 72];  // 128 rows x 64 K (+8 pad)
    __shared__ __align__(16) bf16_t sB[64 * 72];   // 64 rows  x 64 K (+8 pad)
    int tid  = threadIdx.x;
    int lane = tid & 31, wave = tid >> 5;
    int wm = wave & 3, wn = wave >> 2;             // 4(M) x 2(N) waves
    int mbase = blockIdx.y * 128;
    int nbase = blockIdx.x * 64;

    // loaders: A -> 32 bf16/thread (4x uint4), B -> 16 bf16/thread (2x uint4)
    int ar = tid >> 1,  ac = (tid & 1) * 32;
    int br = tid >> 2,  bc = (tid & 3) * 16;
    bool am = (mbase + ar) < M;
    bool bm = (nbase + br) < N;
    const bf16_t* ap = A + (size_t)(mbase + ar) * lda + ac;
    const bf16_t* bp = W + (size_t)(nbase + br) * ldw + bc;

    uint4 areg[4], breg[2];
    auto loadA = [&](int k0){
        if (am){
            #pragma unroll
            for (int i = 0; i < 4; i++) areg[i] = *(const uint4*)(ap + k0 + i*8);
            if (k0 + 128 < K) __builtin_prefetch(ap + k0 + 128, 0, 1);
        } else {
            #pragma unroll
            for (int i = 0; i < 4; i++) areg[i] = make_uint4(0,0,0,0);
        }
    };
    auto loadB = [&](int k0){
        if (bm){
            #pragma unroll
            for (int i = 0; i < 2; i++) breg[i] = *(const uint4*)(bp + k0 + i*8);
            if (k0 + 128 < K) __builtin_prefetch(bp + k0 + 128, 0, 1);
        } else {
            #pragma unroll
            for (int i = 0; i < 2; i++) breg[i] = make_uint4(0,0,0,0);
        }
    };

    v8f acc[2][2] = {};
    loadA(0); loadB(0);

    for (int k0 = 0; k0 < K; k0 += 64) {
        __syncthreads();
        #pragma unroll
        for (int i = 0; i < 4; i++) *(uint4*)(sA + ar * 72 + ac + i*8) = areg[i];
        #pragma unroll
        for (int i = 0; i < 2; i++) *(uint4*)(sB + br * 72 + bc + i*8) = breg[i];
        __syncthreads();
        if (k0 + 64 < K){ loadA(k0 + 64); loadB(k0 + 64); }  // overlap w/ compute

        int hl = lane >> 4;          // lane half -> K sub-range
        int ml = lane & 15;
        #pragma unroll
        for (int ks = 0; ks < 64; ks += 32) {
            v16bf af[2], bf[2];
            #pragma unroll
            for (int mt = 0; mt < 2; ++mt) {
                const bf16_t* arow = sA + (wm*32 + mt*16 + ml) * 72 + ks;
                #pragma unroll
                for (int j = 0; j < 8; ++j) {            // 16-bit A 16x32 layout
                    int kk = ((j < 4) ? 0 : 16) + (j & 3)*2 + hl*8;
                    af[mt][2*j]   = arow[kk];
                    af[mt][2*j+1] = arow[kk+1];
                }
            }
            #pragma unroll
            for (int nt = 0; nt < 2; ++nt) {
                const bf16_t* brow = sB + (wn*32 + nt*16 + ml) * 72 + ks;
                #pragma unroll
                for (int j = 0; j < 8; ++j) {
                    int kk = ((j < 4) ? 0 : 16) + (j & 3)*2 + hl*8;
                    bf[nt][2*j]   = brow[kk];
                    bf[nt][2*j+1] = brow[kk+1];
                }
            }
            #pragma unroll
            for (int mt = 0; mt < 2; ++mt)
                #pragma unroll
                for (int nt = 0; nt < 2; ++nt)
                    acc[mt][nt] = __builtin_amdgcn_wmma_f32_16x16x32_bf16(
                        false, af[mt], false, bf[nt], (short)0, acc[mt][nt],
                        false, false);
        }
    }

    int hl = lane >> 4, nl = lane & 15;
    #pragma unroll
    for (int mt = 0; mt < 2; ++mt)
    #pragma unroll
    for (int nt = 0; nt < 2; ++nt) {
        int gn = nbase + wn*32 + nt*16 + nl;
        #pragma unroll
        for (int j = 0; j < 8; ++j) {                    // 32-bit C/D 16x16 layout
            int gm = mbase + wm*32 + mt*16 + j + hl*8;
            if (gm < M && gn < N) {
                size_t idx = (size_t)gm * ldc + gn;
                float v = acc[mt][nt][j];
                C[idx] = residual ? (C[idx] + v) : v;
            }
        }
    }
}

// ---------------- causal depthwise conv (k=4) + SiLU ----------------
__global__ void k_conv(const float* __restrict__ zx, const float* __restrict__ cw,
                       const float* __restrict__ cb, float* __restrict__ xc){
    long long idx = (long long)blockIdx.x * 256 + threadIdx.x;
    if (idx >= (long long)MTOK * CDIM) return;
    int c   = (int)(idx % CDIM);
    int tok = (int)(idx / CDIM);
    int t = tok % SEQL, b = tok / SEQL;
    float a = cb[c];
    #pragma unroll
    for (int j = 0; j < 4; j++){
        int tt = t - 3 + j;
        if (tt >= 0) a += zx[(size_t)(b * SEQL + tt) * DIP + DI + c] * cw[c * 4 + j];
    }
    xc[(size_t)tok * CDIM + c] = siluf(a);
}

// ---------------- dt = softplus(raw + bias) ----------------
__global__ void k_dt(const float* __restrict__ zx, const float* __restrict__ dtb,
                     float* __restrict__ dt){
    int idx = blockIdx.x * 256 + threadIdx.x;
    if (idx >= MTOK * NH) return;
    int h = idx & 15, tok = idx >> 4;
    float v = zx[(size_t)tok * DIP + (DI + CDIM) + h] + dtb[h];
    dt[idx] = (v > 20.f) ? v : log1pf(__expf(v));
}

// ---------------- sequential SSM scan: 1 block per (b,h), state in regs ----------------
__global__ __launch_bounds__(256)
void k_scan(const float* __restrict__ xc, const float* __restrict__ dt,
            const float* __restrict__ Alog, const float* __restrict__ Dp,
            float* __restrict__ y){
    __shared__ float sB[DS], sC[DS];
    int bh = blockIdx.x;                 // 0..31
    int b = bh >> 4, h = bh & 15;
    int tid = threadIdx.x;
    int p = tid >> 1, n0 = (tid & 1) * 32;   // 2 lanes per p, 32 n each
    float Ah = -__expf(Alog[h]);
    float Dh = Dp[h];
    float st[32];
    #pragma unroll
    for (int i = 0; i < 32; i++) st[i] = 0.f;
    for (int t = 0; t < SEQL; ++t){
        size_t base = (size_t)(b * SEQL + t) * CDIM;
        __syncthreads();
        if (tid < DS)          sB[tid]      = xc[base + DI + tid];
        else if (tid < 2*DS)   sC[tid - DS] = xc[base + DI + DS + (tid - DS)];
        __syncthreads();
        float dtv  = dt[(size_t)(b * SEQL + t) * NH + h];
        float xp   = xc[base + h * HD + p];
        float dA   = __expf(dtv * Ah);
        float coef = dtv * xp;
        float part = 0.f;
        #pragma unroll
        for (int i = 0; i < 32; i++){
            st[i] = st[i] * dA + coef * sB[n0 + i];
            part += st[i] * sC[n0 + i];
        }
        float ysum = part + __shfl_xor(part, 1, 32);   // combine the two n-halves
        if ((tid & 1) == 0)
            y[(size_t)(b * SEQL + t) * DI + h * HD + p] = ysum + xp * Dh;
    }
}

// ---------------- gating * SiLU(z) + RMSNorm -> bf16 ----------------
__global__ void k_gate(const float* __restrict__ yb, const float* __restrict__ zx,
                       const float* __restrict__ rw, bf16_t* __restrict__ y16){
    __shared__ float red[256];
    int tok = blockIdx.x;
    float v[8]; float s = 0.f;
    #pragma unroll
    for (int i = 0; i < 8; i++){
        int e = threadIdx.x + i * 256;
        float z   = zx[(size_t)tok * DIP + e];
        float val = yb[(size_t)tok * DI + e] * siluf(z);
        v[i] = val; s += val * val;
    }
    red[threadIdx.x] = s; __syncthreads();
    for (int o = 128; o > 0; o >>= 1){ if (threadIdx.x < o) red[threadIdx.x] += red[threadIdx.x+o]; __syncthreads(); }
    float rs = rsqrtf(red[0] / DI + 1e-5f);
    #pragma unroll
    for (int i = 0; i < 8; i++){
        int e = threadIdx.x + i * 256;
        y16[(size_t)tok * DI + e] = (bf16_t)(v[i] * rs * rw[e]);
    }
}

// ---------------- head: to_rgb + MSE loss ----------------
__global__ void k_zero(float* p){ if (threadIdx.x == 0 && blockIdx.x == 0) p[0] = 0.f; }

__global__ void k_head(const float* __restrict__ x, const float* __restrict__ tw,
                       const float* __restrict__ tb, const float* __restrict__ im64,
                       float* __restrict__ out){
    __shared__ float r0[256], r1[256], r2[256];
    int bq = blockIdx.x;                       // 0..8191
    int b = bq >> 12, q = bq & 4095;
    const float* xr = x + ((size_t)(b * SEQL) + 66 + q) * DM;
    float a0 = 0.f, a1 = 0.f, a2 = 0.f;
    for (int d = threadIdx.x; d < DM; d += 256){
        float xv = xr[d];
        a0 += xv * tw[d*3+0]; a1 += xv * tw[d*3+1]; a2 += xv * tw[d*3+2];
    }
    r0[threadIdx.x] = a0; r1[threadIdx.x] = a1; r2[threadIdx.x] = a2; __syncthreads();
    for (int o = 128; o > 0; o >>= 1){
        if (threadIdx.x < o){
            r0[threadIdx.x] += r0[threadIdx.x+o];
            r1[threadIdx.x] += r1[threadIdx.x+o];
            r2[threadIdx.x] += r2[threadIdx.x+o];
        }
        __syncthreads();
    }
    if (threadIdx.x == 0){
        float y0 = r0[0] + tb[0], y1 = r1[0] + tb[1], y2 = r2[0] + tb[2];
        size_t o = (size_t)bq * 3;
        out[o] = y0; out[o+1] = y1; out[o+2] = y2;
        float d0 = y0 - im64[o], d1 = y1 - im64[o+1], d2 = y2 - im64[o+2];
        atomicAdd(out + (size_t)BSZ * 4096 * 3,
                  (d0*d0 + d1*d1 + d2*d2) * (1.f / ((float)BSZ * 4096 * 3)));
    }
}

// ---------------- launcher ----------------
extern "C" void kernel_launch(void* const* d_in, const int* in_sizes, int n_in,
                              void* d_out, int out_size, void* d_ws, size_t ws_size,
                              hipStream_t stream) {
    const float* im8   = (const float*)d_in[0];
    const float* im64  = (const float*)d_in[1];
    const float* frw   = (const float*)d_in[2];
    const float* frb   = (const float*)d_in[3];
    const float* trw   = (const float*)d_in[4];
    const float* trb   = (const float*)d_in[5];
    const float* s0    = (const float*)d_in[6];
    const float* s1    = (const float*)d_in[7];
    const float* pe    = (const float*)d_in[8];
    const float* ln_w  = (const float*)d_in[9];
    const float* ln_b  = (const float*)d_in[10];
    const float* ipw   = (const float*)d_in[11];
    const float* cw    = (const float*)d_in[12];
    const float* cb    = (const float*)d_in[13];
    const float* dtb   = (const float*)d_in[14];
    const float* Alog  = (const float*)d_in[15];
    const float* Dvec  = (const float*)d_in[16];
    const float* rmsw  = (const float*)d_in[17];
    const float* opw   = (const float*)d_in[18];
    float* out = (float*)d_out;

    // workspace carve-out
    size_t off = 0;
    auto take = [&](size_t bytes) -> void* {
        void* p = (char*)d_ws + off;
        off += (bytes + 255) & ~(size_t)255;
        return p;
    };
    float*  x     = (float*)  take((size_t)MTOK * DM   * 4);
    bf16_t* u16   = (bf16_t*) take((size_t)MTOK * DM   * 2);
    float*  zx    = (float*)  take((size_t)MTOK * DIP  * 4);
    float*  xc    = (float*)  take((size_t)MTOK * CDIM * 4);
    float*  dts   = (float*)  take((size_t)MTOK * NH   * 4);
    float*  yb    = (float*)  take((size_t)MTOK * DI   * 4);
    bf16_t* y16   = (bf16_t*) take((size_t)MTOK * DI   * 2);
    bf16_t* wi16  = (bf16_t*) take((size_t)DIP * DM    * 2);
    bf16_t* wo16  = (bf16_t*) take((size_t)DM  * DI    * 2);
    (void)ws_size; (void)in_sizes; (void)n_in; (void)out_size;

    k_embed<<<MTOK, 256, 0, stream>>>(im8, frw, frb, s0, s1, pe, x);

    for (int l = 0; l < NL; ++l) {
        k_ln<<<MTOK, 256, 0, stream>>>(x, ln_w + l * DM, ln_b + l * DM, u16);

        int ni = (int)(((size_t)DIP * DM + 255) / 256);
        k_f2bf<<<ni, 256, 0, stream>>>(ipw + (size_t)l * DIP * DM, wi16, DIP * DM);
        dim3 g1((DIP + 63) / 64, (MTOK + 127) / 128);
        k_gemm_bf16<<<g1, 256, 0, stream>>>(u16, DM, wi16, DM, zx, DIP,
                                            MTOK, DIP, DM, 0);

        long long ce = (long long)MTOK * CDIM;
        k_conv<<<(int)((ce + 255) / 256), 256, 0, stream>>>(
            zx, cw + (size_t)l * CDIM * 4, cb + (size_t)l * CDIM, xc);
        k_dt<<<(MTOK * NH + 255) / 256, 256, 0, stream>>>(zx, dtb + l * NH, dts);

        k_scan<<<BSZ * NH, 256, 0, stream>>>(xc, dts, Alog + l * NH, Dvec + l * NH, yb);
        k_gate<<<MTOK, 256, 0, stream>>>(yb, zx, rmsw + (size_t)l * DI, y16);

        int no = (int)(((size_t)DM * DI + 255) / 256);
        k_f2bf<<<no, 256, 0, stream>>>(opw + (size_t)l * DM * DI, wo16, DM * DI);
        dim3 g2((DM + 63) / 64, (MTOK + 127) / 128);
        k_gemm_bf16<<<g2, 256, 0, stream>>>(y16, DI, wo16, DI, x, DM,
                                            MTOK, DM, DI, 1);
    }

    k_zero<<<1, 32, 0, stream>>>(out + (size_t)BSZ * 4096 * 3);
    k_head<<<BSZ * 4096, 256, 0, stream>>>(x, trw, trb, im64, out);
}
